// CausalAttention_35923106464149
// MI455X (gfx1250) — matrix operand order repaired and checked by reading, not compiled, not run on hardware
//
#include <hip/hip_runtime.h>
#include <math.h>

// ---------------------------------------------------------------------------
// Causal attention fwd for B=4, S=2048, D=1024 on gfx1250 (MI455X).
// Compute-bound (~120 GFLOP matmul vs ~100MB HBM traffic) -> everything goes
// through v_wmma_f32_16x16x32_bf16. Intermediates live in d_ws (~166MB),
// mostly resident in the 192MB L2.
//
// GEMM: block 256 thr = 8 wave32 (4x2), block tile 128x128, wave tile 32x64,
// 8 WMMA per 32-deep k-step. Explicit ping-pong double buffering (no register
// copies, no WMMA hazard NOPs) + global_prefetch_b8 256B (4 k-steps) ahead.
// ---------------------------------------------------------------------------

typedef __attribute__((ext_vector_type(16))) __bf16 v16bf;
typedef __attribute__((ext_vector_type(8)))  __bf16 v8bf;
typedef __attribute__((ext_vector_type(8)))  float  v8f;

#define BB 4
#define SS 2048
#define DD 1024

// ---------------------------------------------------------------- conversions
__global__ __launch_bounds__(256)
void cvt_f32_bf16(const float* __restrict__ in, __bf16* __restrict__ out, int n)
{
    int i = blockIdx.x * 256 + threadIdx.x;
    if (i < n) out[i] = (__bf16)in[i];
}

// ---------------------------------------------------------------- fragments
// A frag (16x32 bf16): lane<16 -> M=lane, K=0..15;
//                      lane>=16 -> same M, K=16..31
__device__ __forceinline__ v16bf load_a(const __bf16* p)
{
    v8bf lo = *(const v8bf*)(p);
    v8bf hi = *(const v8bf*)(p + 8);
    return __builtin_shufflevector(lo, hi,
               0,1,2,3,4,5,6,7,8,9,10,11,12,13,14,15);
}
// B frag (32x16 bf16): lane<16 -> N=lane, K=0..15; lane>=16 -> K=16..31
__device__ __forceinline__ v16bf load_b(const __bf16* p)
{
    return *(const v16bf*)(p);
}

#define WMMA_BF16(A_, B_, C_) \
    __builtin_amdgcn_wmma_f32_16x16x32_bf16(false, (A_), false, (B_), \
                                            (short)0, (C_), false, false)

#define WMMA_TILE(A0_, A1_, B0_, B1_, B2_, B3_)        \
    do {                                               \
        acc[0][0] = WMMA_BF16(A0_, B0_, acc[0][0]);    \
        acc[0][1] = WMMA_BF16(A0_, B1_, acc[0][1]);    \
        acc[0][2] = WMMA_BF16(A0_, B2_, acc[0][2]);    \
        acc[0][3] = WMMA_BF16(A0_, B3_, acc[0][3]);    \
        acc[1][0] = WMMA_BF16(A1_, B0_, acc[1][0]);    \
        acc[1][1] = WMMA_BF16(A1_, B1_, acc[1][1]);    \
        acc[1][2] = WMMA_BF16(A1_, B2_, acc[1][2]);    \
        acc[1][3] = WMMA_BF16(A1_, B3_, acc[1][3]);    \
    } while (0)

// ---------------------------------------------------------------- NT GEMM
// C(M,N) = A(M,K) * Bmat(N,K)^T   (A, Bmat bf16 row-major; accum f32)
//   mode 0: store f32 C[row*ldc+col]
//   mode 1: store bf16 C[row*ldc+col]
//   mode 2: store bf16 TRANSPOSED per batch (rows span B*S, col in [0,DD)):
//           C[b*DD*SS + col*SS + s]   (emits V^T directly)
//   causal : waves whose 32x64 tile is fully above the diagonal exit early
//   kclamp : clamp K loop to round32(m0+32) (P zero above diagonal)
__global__ __launch_bounds__(256)
void gemm_nt_bf16(const __bf16* __restrict__ A, int lda, long long strideA,
                  const __bf16* __restrict__ Bm, int ldb, long long strideB,
                  void* __restrict__ Cout, int ldc, long long strideC,
                  int K, int mode, int causal, int kclamp)
{
    const int z = blockIdx.z;
    A  += (long long)z * strideA;
    Bm += (long long)z * strideB;

    const int lane = threadIdx.x & 31;
    const int wave = threadIdx.x >> 5;
    const int wm   = wave >> 1;          // 0..3
    const int wn   = wave & 1;           // 0..1
    const int mblk = blockIdx.y * 128;
    const int m0   = mblk + wm * 32;
    const int n0   = blockIdx.x * 128 + wn * 64;
    if (causal && n0 >= m0 + 32) return;     // wave tile fully above diagonal

    const int lh  = lane >> 4;
    const int l15 = lane & 15;

    const __bf16* arow0 = A + (long long)(m0 + l15) * lda + lh * 16;
    const __bf16* arow1 = arow0 + (long long)16 * lda;
    const __bf16* bp0 = Bm + (long long)(n0 + l15) * ldb + lh * 16;
    const __bf16* bp1 = bp0 + (long long)16 * ldb;
    const __bf16* bp2 = bp0 + (long long)32 * ldb;
    const __bf16* bp3 = bp0 + (long long)48 * ldb;

    int Keff = K;
    if (kclamp) { int kc = (m0 + 32 + 31) & ~31; Keff = kc < K ? kc : K; }
    const int nk = Keff >> 5;            // number of 32-deep k-steps (>= 1)

    v8f acc[2][4] = {};

    // ping-pong fragment sets: P holds the next step to consume, Q the other
    v16bf a0p = load_a(arow0), a1p = load_a(arow1);
    v16bf b0p = load_b(bp0),  b1p = load_b(bp1);
    v16bf b2p = load_b(bp2),  b3p = load_b(bp3);
    v16bf a0q, a1q, b0q, b1q, b2q, b3q;

    int i = 1;
    for (; i + 1 < nk; i += 2) {
        const int k0 = i << 5, k1 = (i + 1) << 5;
        // load Q <- step i (independent of WMMAs on P below)
        a0q = load_a(arow0 + k0); a1q = load_a(arow1 + k0);
        b0q = load_b(bp0 + k0);  b1q = load_b(bp1 + k0);
        b2q = load_b(bp2 + k0);  b3q = load_b(bp3 + k0);
        // 256B (128 bf16 = 4 k-steps) ahead of the Q loads
        __builtin_prefetch(arow0 + k0 + 128, 0, 1);
        __builtin_prefetch(arow1 + k0 + 128, 0, 1);
        __builtin_prefetch(bp0 + k0 + 128, 0, 1);
        __builtin_prefetch(bp1 + k0 + 128, 0, 1);
        __builtin_prefetch(bp2 + k0 + 128, 0, 1);
        __builtin_prefetch(bp3 + k0 + 128, 0, 1);
        WMMA_TILE(a0p, a1p, b0p, b1p, b2p, b3p);       // consume step i-1
        // load P <- step i+1
        a0p = load_a(arow0 + k1); a1p = load_a(arow1 + k1);
        b0p = load_b(bp0 + k1);  b1p = load_b(bp1 + k1);
        b2p = load_b(bp2 + k1);  b3p = load_b(bp3 + k1);
        WMMA_TILE(a0q, a1q, b0q, b1q, b2q, b3q);       // consume step i
    }
    if (i < nk) {                         // one leftover k-step to load
        const int k0 = i << 5;
        a0q = load_a(arow0 + k0); a1q = load_a(arow1 + k0);
        b0q = load_b(bp0 + k0);  b1q = load_b(bp1 + k0);
        b2q = load_b(bp2 + k0);  b3q = load_b(bp3 + k0);
        WMMA_TILE(a0p, a1p, b0p, b1p, b2p, b3p);
        WMMA_TILE(a0q, a1q, b0q, b1q, b2q, b3q);
    } else {
        WMMA_TILE(a0p, a1p, b0p, b1p, b2p, b3p);
    }

    // C/D layout: lane<16 -> row = base+i, lane>=16 -> row = base+8+i; col=l15
    if (mode == 0) {
        float* C = (float*)Cout + (long long)z * strideC;
        #pragma unroll
        for (int mi = 0; mi < 2; ++mi)
            #pragma unroll
            for (int j = 0; j < 4; ++j)
                #pragma unroll
                for (int r = 0; r < 8; ++r) {
                    int row = m0 + mi * 16 + lh * 8 + r;
                    int col = n0 + j * 16 + l15;
                    C[(long long)row * ldc + col] = acc[mi][j][r];
                }
    } else if (mode == 1) {
        __bf16* C = (__bf16*)Cout + (long long)z * strideC;
        #pragma unroll
        for (int mi = 0; mi < 2; ++mi)
            #pragma unroll
            for (int j = 0; j < 4; ++j)
                #pragma unroll
                for (int r = 0; r < 8; ++r) {
                    int row = m0 + mi * 16 + lh * 8 + r;
                    int col = n0 + j * 16 + l15;
                    C[(long long)row * ldc + col] = (__bf16)acc[mi][j][r];
                }
    } else {
        __bf16* C = (__bf16*)Cout;   // V^T: [b][DD][SS]
        #pragma unroll
        for (int mi = 0; mi < 2; ++mi)
            #pragma unroll
            for (int j = 0; j < 4; ++j)
                #pragma unroll
                for (int r = 0; r < 8; ++r) {
                    int row = m0 + mi * 16 + lh * 8 + r;   // global b*SS + s
                    int col = n0 + j * 16 + l15;           // d
                    int b   = row >> 11;                   // SS = 2048
                    int s   = row & (SS - 1);
                    C[(long long)b * DD * SS + (long long)col * SS + s] =
                        (__bf16)acc[mi][j][r];
                }
    }
}

// ---------------------------------------------------------------- softmax
// One 256-thread block per (q,b) row: scale, causal mask, max, exp, sum,
// normalize, emit bf16 P. 2048 = 256 threads * 8 elements.
__global__ __launch_bounds__(256)
void softmax_causal(const float* __restrict__ Sc, __bf16* __restrict__ P,
                    float scale)
{
    const int q = blockIdx.x;
    const int b = blockIdx.y;
    const float* row  = Sc + ((long long)b * SS + q) * SS;
    __bf16*      prow = P  + ((long long)b * SS + q) * SS;
    const int t = threadIdx.x;

    float v[8];
    float m = -__builtin_inff();
    #pragma unroll
    for (int i = 0; i < 8; ++i) {
        int j = t + i * 256;
        float x = (j <= q) ? row[j] * scale : -__builtin_inff();
        v[i] = x;
        m = fmaxf(m, x);
    }

    __shared__ float red[8];
    #pragma unroll
    for (int o = 16; o > 0; o >>= 1) m = fmaxf(m, __shfl_xor(m, o, 32));
    if ((t & 31) == 0) red[t >> 5] = m;
    __syncthreads();
    m = red[0];
    #pragma unroll
    for (int i = 1; i < 8; ++i) m = fmaxf(m, red[i]);
    __syncthreads();

    float s = 0.f;
    #pragma unroll
    for (int i = 0; i < 8; ++i) {
        float e = __expf(v[i] - m);   // exp(-inf) -> 0 handles masked lanes
        v[i] = e;
        s += e;
    }
    #pragma unroll
    for (int o = 16; o > 0; o >>= 1) s += __shfl_xor(s, o, 32);
    if ((t & 31) == 0) red[t >> 5] = s;
    __syncthreads();
    s = 0.f;
    #pragma unroll
    for (int i = 0; i < 8; ++i) s += red[i];

    float inv = 1.0f / s;
    #pragma unroll
    for (int i = 0; i < 8; ++i)
        prow[t + i * 256] = (__bf16)(v[i] * inv);
}

// ---------------------------------------------------------------- launcher
extern "C" void kernel_launch(void* const* d_in, const int* in_sizes, int n_in,
                              void* d_out, int out_size, void* d_ws,
                              size_t ws_size, hipStream_t stream)
{
    const float* x  = (const float*)d_in[0];
    const float* Wq = (const float*)d_in[1];
    const float* Wk = (const float*)d_in[2];
    const float* Wv = (const float*)d_in[3];
    float* out = (float*)d_out;

    // workspace layout (bytes): all offsets multiples of 2MB
    char* ws = (char*)d_ws;
    const long long NX = (long long)BB * SS * DD;   // 8,388,608
    const long long NW = (long long)DD * DD;        // 1,048,576
    size_t off = 0;
    __bf16* xh  = (__bf16*)(ws + off); off += NX * 2;          // 16MB
    __bf16* wqh = (__bf16*)(ws + off); off += NW * 2;          //  2MB
    __bf16* wkh = (__bf16*)(ws + off); off += NW * 2;
    __bf16* wvh = (__bf16*)(ws + off); off += NW * 2;
    __bf16* qh  = (__bf16*)(ws + off); off += NX * 2;          // 16MB
    __bf16* kh  = (__bf16*)(ws + off); off += NX * 2;          // 16MB
    __bf16* vTh = (__bf16*)(ws + off); off += NX * 2;          // 16MB (V^T)
    float*  sc  = (float*) (ws + off); off += (long long)BB * SS * SS * 4; // 64MB
    __bf16* P   = (__bf16*)(ws + off); off += (long long)BB * SS * SS * 2; // 32MB

    // 1) f32 -> bf16
    cvt_f32_bf16<<<(int)(NX / 256), 256, 0, stream>>>(x,  xh,  (int)NX);
    cvt_f32_bf16<<<(int)(NW / 256), 256, 0, stream>>>(Wq, wqh, (int)NW);
    cvt_f32_bf16<<<(int)(NW / 256), 256, 0, stream>>>(Wk, wkh, (int)NW);
    cvt_f32_bf16<<<(int)(NW / 256), 256, 0, stream>>>(Wv, wvh, (int)NW);

    dim3 blk(256);

    // 2) QKV projections: M = B*S = 8192, N = D, K = D  (q = x * Wq^T ...)
    dim3 gq(DD / 128, (BB * SS) / 128, 1);
    gemm_nt_bf16<<<gq, blk, 0, stream>>>(xh, DD, 0, wqh, DD, 0,
                                         qh,  DD, 0, DD, 1, 0, 0);
    gemm_nt_bf16<<<gq, blk, 0, stream>>>(xh, DD, 0, wkh, DD, 0,
                                         kh,  DD, 0, DD, 1, 0, 0);
    gemm_nt_bf16<<<gq, blk, 0, stream>>>(xh, DD, 0, wvh, DD, 0,
                                         vTh, DD, 0, DD, 2, 0, 0);  // -> V^T

    // 3) scores = Q * K^T per batch (causal wave-tile skip), f32 out
    dim3 gs(SS / 128, SS / 128, BB);
    gemm_nt_bf16<<<gs, blk, 0, stream>>>(
        qh, DD, (long long)SS * DD,
        kh, DD, (long long)SS * DD,
        sc, SS, (long long)SS * SS,
        DD, 0, 1, 0);

    // 4) causal softmax (scale = 1/sqrt(1024) = 1/32), emit bf16 P
    softmax_causal<<<dim3(SS, BB), blk, 0, stream>>>(sc, P, 1.0f / 32.0f);

    // 5) out = P * V  (== P * (V^T)^T, NT form), K clamped to diagonal
    dim3 go(DD / 128, SS / 128, BB);
    gemm_nt_bf16<<<go, blk, 0, stream>>>(
        P,   SS, (long long)SS * SS,
        vTh, SS, (long long)DD * SS,
        out, DD, (long long)SS * DD,
        SS, 0, 0, 1);
}
